// ACTOR_34935263986357
// MI455X (gfx1250) — matrix operand order repaired and checked by reading, not compiled, not run on hardware
//
#include <hip/hip_runtime.h>
#include <hip/hip_bf16.h>
#include <math.h>

#define Ee   128
#define Hh   8
#define KDd  16
#define FFf  512
#define Ll   6
#define Nn   200
#define Ss   50
#define Gg   100
#define Bb   128
#define Pp   250           // Nn + Ss
#define TPd  288           // padded token count for transposed V (>= kOff + keyStripMax)
#define CLIPV 10.0f
#define EPSV  1e-5f
#define SCALE_KD 0.25f                 // 1/sqrt(16)
#define INV_SQ_E 0.08838834764831845f  // 1/sqrt(128)
#define LKP 224                        // max padded key strip (ceil(200/32)*32)

typedef __bf16 bf16t;
typedef __attribute__((ext_vector_type(16))) bf16t v16bf;
typedef __attribute__((ext_vector_type(8)))  bf16t v8bf;
typedef __attribute__((ext_vector_type(8)))  float v8f;

__device__ __forceinline__ v8f wmma_bf16(v16bf a, v16bf b, v8f c) {
  // D = A(16x32 bf16) x B(32x16 bf16) + C(16x16 f32)
  return __builtin_amdgcn_wmma_f32_16x16x32_bf16(false, a, false, b, (short)0, c, false, false);
}

// Convert 8 consecutive floats (two aligned float4s) into bf16 elements [o..o+7]
__device__ __forceinline__ void cvt8(v16bf& dst, int o, const float* __restrict__ p) {
  const float4 f0 = *(const float4*)p;
  const float4 f1 = *(const float4*)(p + 4);
  dst[o+0] = (bf16t)f0.x; dst[o+1] = (bf16t)f0.y; dst[o+2] = (bf16t)f0.z; dst[o+3] = (bf16t)f0.w;
  dst[o+4] = (bf16t)f1.x; dst[o+5] = (bf16t)f1.y; dst[o+6] = (bf16t)f1.z; dst[o+7] = (bf16t)f1.w;
}

// ---------------------------------------------------------------------------
// Pack fp32 weight (K x Nc, row major) into bf16 WMMA B-fragment order:
//   Wp[kblk][ncol][kk], kblk=k/32, kk in table order (kk 8-groups hold k%32 groups 0,2,1,3). One v8bf store per thread.
// ---------------------------------------------------------------------------
__global__ __launch_bounds__(256) void packW_kernel(
    const float* __restrict__ W, bf16t* __restrict__ Wp, int K, int Nc)
{
  int idx = blockIdx.x * 256 + threadIdx.x;
  if (idx >= (K * Nc) / 8) return;
  int kk8  = idx & 3;         // which group of 8 within the 32-k block
  int t    = idx >> 2;
  int ncol = t % Nc;
  int kblk = t / Nc;
  v8bf o;
#pragma unroll
  for (int j = 0; j < 8; ++j)
    o[j] = (bf16t)W[(size_t)(kblk * 32 + ((((kk8 & 1) << 1) | (kk8 >> 1)) * 8) + j) * Nc + ncol];
  *(v8bf*)(Wp + ((size_t)kblk * Nc + ncol) * 32 + kk8 * 8) = o;
}

// ---------------------------------------------------------------------------
// Transpose V (B,P,E) fp32 -> Vt (B,E,TPd) bf16, zero-padded past P tokens.
// Each thread emits one contiguous v8bf (8 tokens).
// ---------------------------------------------------------------------------
__global__ __launch_bounds__(256) void vtrans_kernel(
    const float* __restrict__ V, bf16t* __restrict__ Vt)
{
  int idx = blockIdx.x * 256 + threadIdx.x;
  if (idx >= Bb * Ee * (TPd / 8)) return;
  int t8 = idx % (TPd / 8);
  int r  = idx / (TPd / 8);   // b*Ee + e
  int e = r % Ee;
  int b = r / Ee;
  int t0 = t8 * 8;
  v8bf o;
#pragma unroll
  for (int j = 0; j < 8; ++j) {
    int t = t0 + j;
    float v = (t < Pp) ? V[((size_t)b * Pp + t) * Ee + e] : 0.f;
    o[j] = (bf16t)v;
  }
  *(v8bf*)(Vt + (size_t)r * TPd + t0) = o;
}

// ---------------------------------------------------------------------------
// Embedding
// ---------------------------------------------------------------------------
__global__ __launch_bounds__(256) void embed_kernel(
    const float* __restrict__ item, const float* __restrict__ sols,
    const float* __restrict__ embW, const float* __restrict__ embB,
    const float* __restrict__ sembW, const float* __restrict__ sembB,
    float* __restrict__ X)
{
  int idx = blockIdx.x * 256 + threadIdx.x;
  if (idx >= Bb * Pp * Ee) return;
  int e = idx % Ee;
  int t = idx / Ee;
  int p = t % Pp;
  int b = t / Pp;
  float v;
  if (p < Nn) {
    const float* it = item + ((size_t)b * Nn + p) * 3;
    v = it[0]*embW[e] + it[1]*embW[Ee+e] + it[2]*embW[2*Ee+e] + embB[e];
  } else {
    const float* sp = sols + ((size_t)b * Ss + (p - Nn)) * 2;
    v = sp[0]*sembW[e] + sp[1]*sembW[Ee+e] + sembB[e];
  }
  X[idx] = v;
}

// ---------------------------------------------------------------------------
// WMMA GEMM: C = act(A @ W + bias). Per-wave 32(M) x 64(N) tile, K templated
// (128 or 512) and unrolled x2 for load/WMMA overlap. A fp32 -> bf16 on load,
// W pre-packed bf16 fragments (one 32B vector load per fragment).
// Logical row r maps to token row (r/RN)*bTok + rOff + r%RN.
// ---------------------------------------------------------------------------
template<int KC>
__global__ __launch_bounds__(256) void gemm_kernel(
    const float* __restrict__ A, const bf16t* __restrict__ Wp,
    const float* __restrict__ bias, float* __restrict__ C,
    int M, int Nc, int RN, int abTok, int arOff,
    int cbTok, int crOff, int relu)
{
  int wid  = threadIdx.x >> 5;
  int lane = threadIdx.x & 31;
  int nTiles = Nc >> 6;
  int w = blockIdx.x * 8 + wid;
  int totalWaves = (M >> 5) * nTiles;
  if (w >= totalWaves) return;
  int nt = w % nTiles;
  int mt = w / nTiles;
  int m    = lane & 15;
  int half = lane >> 4;

  int r0 = mt * 32 + m;
  int r1 = r0 + 16;
  int b0 = r0 / RN, b1 = r1 / RN;
  const float* arow0 = A + ((size_t)b0 * abTok + arOff + (r0 - b0 * RN)) * (size_t)KC;
  const float* arow1 = A + ((size_t)b1 * abTok + arOff + (r1 - b1 * RN)) * (size_t)KC;
  int n0 = nt << 6;

  v8f c00={},c01={},c02={},c03={},c10={},c11={},c12={},c13={};
#pragma unroll 2
  for (int k0 = 0; k0 < KC; k0 += 32) {
    v16bf a0, a1;
    cvt8(a0, 0, arow0 + k0 + half * 8);
    cvt8(a0, 8, arow0 + k0 + 16 + half * 8);
    cvt8(a1, 0, arow1 + k0 + half * 8);
    cvt8(a1, 8, arow1 + k0 + 16 + half * 8);
    const bf16t* wb = Wp + ((size_t)(k0 >> 5) * Nc) * 32 + half * 16;
    v16bf bf0 = *(const v16bf*)(wb + (size_t)(n0      + m) * 32);
    v16bf bf1 = *(const v16bf*)(wb + (size_t)(n0 + 16 + m) * 32);
    v16bf bf2 = *(const v16bf*)(wb + (size_t)(n0 + 32 + m) * 32);
    v16bf bf3 = *(const v16bf*)(wb + (size_t)(n0 + 48 + m) * 32);
    c00 = wmma_bf16(a0, bf0, c00);  c01 = wmma_bf16(a0, bf1, c01);
    c02 = wmma_bf16(a0, bf2, c02);  c03 = wmma_bf16(a0, bf3, c03);
    c10 = wmma_bf16(a1, bf0, c10);  c11 = wmma_bf16(a1, bf1, c11);
    c12 = wmma_bf16(a1, bf2, c12);  c13 = wmma_bf16(a1, bf3, c13);
  }

  auto store_tile = [&](v8f accv, int g, int s) {
    int ncol = n0 + s * 16 + m;
    float bv = bias ? bias[ncol] : 0.0f;
#pragma unroll
    for (int r = 0; r < 8; ++r) {
      int rl  = mt * 32 + g * 16 + r + 8 * half;
      int b2  = rl / RN;
      int in2 = rl - b2 * RN;
      float v = accv[r] + bv;
      if (relu) v = fmaxf(v, 0.0f);
      C[((size_t)b2 * cbTok + crOff + in2) * (size_t)Nc + ncol] = v;
    }
  };
  store_tile(c00, 0, 0); store_tile(c01, 0, 1); store_tile(c02, 0, 2); store_tile(c03, 0, 3);
  store_tile(c10, 1, 0); store_tile(c11, 1, 1); store_tile(c12, 1, 2); store_tile(c13, 1, 3);
}

// ---------------------------------------------------------------------------
// Fused MHA: one wave per (b, h, 16-query tile); Lk templated (200 or 50).
// Score strip in private LDS, in-wave softmax, P@V with bf16 WMMA reading
// transposed bf16 V. mask element = mask[b*maskB + row*maskR + key].
// ---------------------------------------------------------------------------
template<int LKt>
__global__ __launch_bounds__(128) void attn_kernel(
    const float* __restrict__ Q, int qbTok, int qOff, int Lq,
    const float* __restrict__ Kmat, int kbTok,
    const bf16t* __restrict__ Vt, int kOff,
    float* __restrict__ O, int obTok, int oOff, int acc,
    const float* __restrict__ mask, int maskB, int maskR,
    int numWaves, int QT)
{
  constexpr int KT32 = (LKt + 31) / 32;
  constexpr int KT16 = KT32 * 2;
  constexpr int LKC  = KT32 * 32;
  __shared__ float sS[4][16 * LKP];
  __shared__ float sSum[4][16];
  int wid  = threadIdx.x >> 5;
  int lane = threadIdx.x & 31;
  int w = blockIdx.x * 4 + wid;
  if (w >= numWaves) return;
  int qt = w % QT;
  int bh = w / QT;
  int h = bh % Hh;
  int b = bh / Hh;
  int m    = lane & 15;
  int half = lane >> 4;
  int qbase = qt << 4;
  float* strip = sS[wid];

  // A fragment: 16 query rows, K = head dims 0..15, dims 16..31 zero-padded.
  int qr = qbase + m; if (qr >= Lq) qr = Lq - 1;
  const float* qp = Q + ((size_t)b * qbTok + qOff + qr) * Ee + h * KDd;
  v16bf aq;
  cvt8(aq, 0, qp + half * 8);
#pragma unroll
  for (int j = 8; j < 16; ++j) aq[j] = (bf16t)0.0f;

  // Scores (q . k^T / sqrt(KD) + mask) -> LDS strip, -inf padding past Lk.
#pragma unroll 2
  for (int j16 = 0; j16 < KT16; ++j16) {
    int kb = j16 << 4;
    v16bf bq;
    {
      int kr = kb + m; if (kr >= LKt) kr = LKt - 1;
      const float* kp = Kmat + ((size_t)b * kbTok + kOff + kr) * Ee + h * KDd;
      cvt8(bq, 0, kp + half * 8);
#pragma unroll
      for (int j = 8; j < 16; ++j) bq[j] = (bf16t)0.0f;
    }
    v8f c = {};
    c = wmma_bf16(aq, bq, c);
    int col = kb + m;
    bool colValid = col < LKt;
#pragma unroll
    for (int r = 0; r < 8; ++r) {
      int row = r + 8 * half;
      float v;
      if (colValid) {
        v = c[r] * SCALE_KD;
        if (mask) {
          int mrow = qbase + row; if (mrow >= Lq) mrow = Lq - 1;
          v += mask[(size_t)b * maskB + (size_t)mrow * maskR + col];
        }
      } else v = -1e30f;
      strip[row * LKP + col] = v;
    }
  }

  // Row softmax (one lane per query row; same-wave LDS traffic -> in order).
  if (lane < 16) {
    float* row = strip + lane * LKP;
    float mx = -1e30f;
#pragma unroll 4
    for (int j = 0; j < LKC; ++j) mx = fmaxf(mx, row[j]);
    float sum = 0.0f;
#pragma unroll 4
    for (int j = 0; j < LKC; ++j) { float e2 = __expf(row[j] - mx); row[j] = e2; sum += e2; }
    sSum[wid][lane] = sum;
  }

  // O = P @ V  (bf16 WMMA over 32-key chunks; Vt rows contiguous along tokens)
  v8f o = {};
  const bf16t* vrow = Vt + ((size_t)b * Ee + h * KDd + m) * TPd + kOff;
  const float* prow = strip + m * LKP;
#pragma unroll 2
  for (int kc = 0; kc < KT32; ++kc) {
    int k0 = kc << 5;
    v16bf ap;
    cvt8(ap, 0, prow + k0 + half * 8);
    cvt8(ap, 8, prow + k0 + 16 + half * 8);
    v16bf bp;
    const v8bf b0 = *(const v8bf*)(vrow + k0 + half * 8);
    const v8bf b1 = *(const v8bf*)(vrow + k0 + 16 + half * 8);
#pragma unroll
    for (int j = 0; j < 8; ++j) { bp[j] = b0[j]; bp[8+j] = b1[j]; }
    o = wmma_bf16(ap, bp, o);
  }

  // Normalize and write (D layout: row = r+8*half, col = lane&15).
#pragma unroll
  for (int r = 0; r < 8; ++r) {
    int row = r + 8 * half;
    int qg = qbase + row;
    if (qg < Lq) {
      float val = o[r] / sSum[wid][row];
      float* op = O + ((size_t)b * obTok + oOff + qg) * Ee + h * KDd + m;
      if (acc) *op += val; else *op = val;
    }
  }
}

// ---------------------------------------------------------------------------
// score[b,g,n] = CLIP*tanh((mh[b,g,:] . shk[b,n,:]) / sqrt(E)) + ninf[b,g,n]
// ---------------------------------------------------------------------------
__global__ __launch_bounds__(128) void score_kernel(
    const float* __restrict__ MHd, const float* __restrict__ SHK,
    const float* __restrict__ ninf, float* __restrict__ Sc,
    int numWaves, int GT, int NT)
{
  int wid  = threadIdx.x >> 5;
  int lane = threadIdx.x & 31;
  int w = blockIdx.x * 4 + wid;
  if (w >= numWaves) return;
  int nt  = w % NT;
  int rem = w / NT;
  int gt  = rem % GT;
  int b   = rem / GT;
  int m    = lane & 15;
  int half = lane >> 4;
  int g = gt * 16 + m; if (g >= Gg) g = Gg - 1;
  const float* arow = MHd + ((size_t)b * Gg + g) * Ee;
  int n = nt * 16 + m; if (n >= Nn) n = Nn - 1;
  const float* brow = SHK + ((size_t)b * Nn + n) * Ee;
  v8f c = {};
#pragma unroll
  for (int k0 = 0; k0 < Ee; k0 += 32) {
    v16bf a, bfr;
    cvt8(a, 0, arow + k0 + half * 8);
    cvt8(a, 8, arow + k0 + 16 + half * 8);
    cvt8(bfr, 0, brow + k0 + half * 8);
    cvt8(bfr, 8, brow + k0 + 16 + half * 8);
    c = wmma_bf16(a, bfr, c);
  }
  int col = nt * 16 + m;
  if (col < Nn) {
#pragma unroll
    for (int r = 0; r < 8; ++r) {
      int gr = gt * 16 + r + 8 * half;
      if (gr < Gg) {
        size_t idx = ((size_t)b * Gg + gr) * Nn + col;
        Sc[idx] = CLIPV * tanhf(c[r] * INV_SQ_E) + ninf[idx];
      }
    }
  }
}

// ---------------------------------------------------------------------------
// BatchNorm stats: 32 blocks, one float4 feature-column each (b128 loads).
// ---------------------------------------------------------------------------
__global__ __launch_bounds__(256) void bnstats_kernel(
    const float* __restrict__ X, float* __restrict__ stats, int rows)
{
  __shared__ float4 rs[256], rq[256];
  int c = blockIdx.x;                      // float4 column 0..31
  const float4* X4 = (const float4*)X;
  float4 s  = {0.f,0.f,0.f,0.f};
  float4 s2 = {0.f,0.f,0.f,0.f};
  for (int r = threadIdx.x; r < rows; r += 256) {
    float4 v = X4[(size_t)r * (Ee/4) + c];
    s.x += v.x; s.y += v.y; s.z += v.z; s.w += v.w;
    s2.x += v.x*v.x; s2.y += v.y*v.y; s2.z += v.z*v.z; s2.w += v.w*v.w;
  }
  rs[threadIdx.x] = s; rq[threadIdx.x] = s2;
  __syncthreads();
  for (int off = 128; off > 0; off >>= 1) {
    if (threadIdx.x < off) {
      float4 a = rs[threadIdx.x], bb = rs[threadIdx.x + off];
      a.x += bb.x; a.y += bb.y; a.z += bb.z; a.w += bb.w;
      rs[threadIdx.x] = a;
      float4 q = rq[threadIdx.x], qq = rq[threadIdx.x + off];
      q.x += qq.x; q.y += qq.y; q.z += qq.z; q.w += qq.w;
      rq[threadIdx.x] = q;
    }
    __syncthreads();
  }
  if (threadIdx.x == 0) {
    int e = c * 4;
    float4 sm = rs[0], sq = rq[0];
    float inv = 1.0f / rows;
    float mean0 = sm.x*inv, mean1 = sm.y*inv, mean2 = sm.z*inv, mean3 = sm.w*inv;
    stats[e+0] = mean0; stats[Ee+e+0] = rsqrtf(sq.x*inv - mean0*mean0 + EPSV);
    stats[e+1] = mean1; stats[Ee+e+1] = rsqrtf(sq.y*inv - mean1*mean1 + EPSV);
    stats[e+2] = mean2; stats[Ee+e+2] = rsqrtf(sq.z*inv - mean2*mean2 + EPSV);
    stats[e+3] = mean3; stats[Ee+e+3] = rsqrtf(sq.w*inv - mean3*mean3 + EPSV);
  }
}

__global__ __launch_bounds__(256) void bnapply_kernel(
    float* __restrict__ X, const float* __restrict__ stats,
    const float* __restrict__ g, const float* __restrict__ bb, int n4)
{
  int idx = blockIdx.x * 256 + threadIdx.x;
  if (idx >= n4) return;
  int e = (idx << 2) & (Ee - 1);
  float4 x  = ((float4*)X)[idx];
  float4 mv = *(const float4*)(stats + e);
  float4 rv = *(const float4*)(stats + Ee + e);
  float4 gv = *(const float4*)(g + e);
  float4 bv = *(const float4*)(bb + e);
  x.x = (x.x - mv.x) * rv.x * gv.x + bv.x;
  x.y = (x.y - mv.y) * rv.y * gv.y + bv.y;
  x.z = (x.z - mv.z) * rv.z * gv.z + bv.z;
  x.w = (x.w - mv.w) * rv.w * gv.w + bv.w;
  ((float4*)X)[idx] = x;
}

__global__ __launch_bounds__(256) void add_kernel(
    const float4* __restrict__ a, const float4* __restrict__ b,
    float4* __restrict__ c, int n4)
{
  int idx = blockIdx.x * 256 + threadIdx.x;
  if (idx >= n4) return;
  float4 x = a[idx], y = b[idx];
  x.x += y.x; x.y += y.y; x.z += y.z; x.w += y.w;
  c[idx] = x;
}

__global__ __launch_bounds__(256) void graphmean_kernel(
    const float* __restrict__ X, float* __restrict__ graph)
{
  int idx = blockIdx.x * 256 + threadIdx.x;
  if (idx >= Bb * (Ee/4)) return;
  int b = idx / (Ee/4), c = idx % (Ee/4);
  const float4* p = (const float4*)X + (size_t)b * Pp * (Ee/4) + c;
  float4 s = {0.f,0.f,0.f,0.f};
  for (int t = 0; t < Pp; ++t) {
    float4 v = p[(size_t)t * (Ee/4)];
    s.x += v.x; s.y += v.y; s.z += v.z; s.w += v.w;
  }
  s.x *= (1.0f/Pp); s.y *= (1.0f/Pp); s.z *= (1.0f/Pp); s.w *= (1.0f/Pp);
  ((float4*)graph)[idx] = s;
}

__global__ __launch_bounds__(256) void qexpand_kernel(
    const float* __restrict__ qb, const float* __restrict__ cap,
    const float* __restrict__ wlast, float* __restrict__ qd)
{
  int idx = blockIdx.x * 256 + threadIdx.x;
  if (idx >= Bb * Gg * (Ee/4)) return;
  int c = idx % (Ee/4);
  int t = idx / (Ee/4);
  int g = t % Gg;
  int b = t / Gg;
  float cv = cap[b * Gg + g];
  float4 q = *((const float4*)qb + (size_t)b * (Ee/4) + c);
  float4 wl = *((const float4*)wlast + c);
  q.x += cv * wl.x; q.y += cv * wl.y; q.z += cv * wl.z; q.w += cv * wl.w;
  ((float4*)qd)[idx] = q;
}

__global__ __launch_bounds__(256) void softmax_kernel(
    const float* __restrict__ Sc, float* __restrict__ out)
{
  __shared__ float red[256];
  int row = blockIdx.x;
  const float* x = Sc + (size_t)row * Nn;
  float mx = -1e30f;
  for (int j = threadIdx.x; j < Nn; j += 256) mx = fmaxf(mx, x[j]);
  red[threadIdx.x] = mx; __syncthreads();
  for (int off = 128; off > 0; off >>= 1) {
    if (threadIdx.x < off) red[threadIdx.x] = fmaxf(red[threadIdx.x], red[threadIdx.x + off]);
    __syncthreads();
  }
  mx = red[0]; __syncthreads();
  float s = 0.f;
  for (int j = threadIdx.x; j < Nn; j += 256) s += __expf(x[j] - mx);
  red[threadIdx.x] = s; __syncthreads();
  for (int off = 128; off > 0; off >>= 1) {
    if (threadIdx.x < off) red[threadIdx.x] += red[threadIdx.x + off];
    __syncthreads();
  }
  s = red[0];
  float inv = 1.0f / s;
  for (int j = threadIdx.x; j < Nn; j += 256)
    out[(size_t)row * Nn + j] = __expf(x[j] - mx) * inv;
}

// ---------------------------------------------------------------------------
// Host orchestration
// ---------------------------------------------------------------------------
extern "C" void kernel_launch(void* const* d_in, const int* in_sizes, int n_in,
                              void* d_out, int out_size, void* d_ws, size_t ws_size,
                              hipStream_t stream) {
  (void)in_sizes; (void)n_in; (void)out_size; (void)ws_size;
  const float* item          = (const float*)d_in[0];
  const float* sols          = (const float*)d_in[1];
  const float* sols_mask     = (const float*)d_in[2];
  const float* capacity      = (const float*)d_in[3];
  const float* sols_mask_pomo= (const float*)d_in[4];
  const float* ninf_mask     = (const float*)d_in[5];
  const float* emb_W   = (const float*)d_in[6];
  const float* emb_b   = (const float*)d_in[7];
  const float* semb_W  = (const float*)d_in[8];
  const float* semb_b  = (const float*)d_in[9];
  const float* LWq     = (const float*)d_in[10];
  const float* LWk     = (const float*)d_in[11];
  const float* LWv     = (const float*)d_in[12];
  const float* Lcomb_W = (const float*)d_in[13];
  const float* Lcomb_b = (const float*)d_in[14];
  const float* Lcombs_W= (const float*)d_in[15];
  const float* Lcombs_b= (const float*)d_in[16];
  const float* Lbn1_g  = (const float*)d_in[17];
  const float* Lbn1_b  = (const float*)d_in[18];
  const float* Lff_W1  = (const float*)d_in[19];
  const float* Lff_b1  = (const float*)d_in[20];
  const float* Lff_W2  = (const float*)d_in[21];
  const float* Lff_b2  = (const float*)d_in[22];
  const float* Lbn2_g  = (const float*)d_in[23];
  const float* Lbn2_b  = (const float*)d_in[24];
  const float* dWq     = (const float*)d_in[25];
  const float* dWk     = (const float*)d_in[26];
  const float* dWv     = (const float*)d_in[27];
  const float* dcomb_W = (const float*)d_in[28];
  const float* dcomb_b = (const float*)d_in[29];
  const float* dWk_logit = (const float*)d_in[30];

  float* ws = (float*)d_ws;
  const size_t SZ = (size_t)Bb * Pp * Ee;          // 4,096,000 floats
  float* X    = ws;
  float* Qb   = X   + SZ;
  float* Kb   = Qb  + SZ;
  float* Vb   = Kb  + SZ;
  float* OC   = Vb  + SZ;
  float* MHb  = OC  + SZ;
  float* T1   = MHb + SZ;
  float* T2   = T1  + SZ;
  float* FFb  = T2  + SZ;                          // B*P*FF floats
  float* stats = FFb + (size_t)Bb * Pp * FFf;      // 256 floats
  bf16t* wpk  = (bf16t*)(stats + 256);             // packed bf16 weights
  // decoder aliases (encoder scratch no longer live)
  float* graphB = T1;
  float* qbase  = T1 + Bb * Ee;
  float* qdec   = Qb;
  float* shk    = FFb;     // (B*N, E)
  float* scoreB = T2;      // (B, G, N)
  float* ocdec  = OC;      // (B, G, E)
  float* mhdec  = MHb;     // (B, G, E)

  const int ROWS = Bb * Pp;        // 32000
  const int NE   = ROWS * Ee;      // 4,096,000
  const int NE4  = NE / 4;

  // ---- pack all weights to bf16 fragment order (tiny, L2-resident) ----
  size_t woff = 0;
  auto packW = [&](const float* W, int K, int Nc) -> bf16t* {
    bf16t* p = wpk + woff;
    int n8 = (K * Nc) / 8;
    packW_kernel<<<(n8 + 255) / 256, 256, 0, stream>>>(W, p, K, Nc);
    woff += (size_t)K * Nc;
    return p;
  };
  bf16t *pWq[Ll], *pWk[Ll], *pWv[Ll], *pCo[Ll], *pCs[Ll], *pF1[Ll], *pF2[Ll];
  for (int i = 0; i < Ll; ++i) {
    pWq[i] = packW(LWq     + (size_t)i * Ee * Ee,  Ee,  Ee);
    pWk[i] = packW(LWk     + (size_t)i * Ee * Ee,  Ee,  Ee);
    pWv[i] = packW(LWv     + (size_t)i * Ee * Ee,  Ee,  Ee);
    pCo[i] = packW(Lcomb_W + (size_t)i * Ee * Ee,  Ee,  Ee);
    pCs[i] = packW(Lcombs_W+ (size_t)i * Ee * Ee,  Ee,  Ee);
    pF1[i] = packW(Lff_W1  + (size_t)i * Ee * FFf, Ee,  FFf);
    pF2[i] = packW(Lff_W2  + (size_t)i * FFf * Ee, FFf, Ee);
  }
  bf16t* pdWk   = packW(dWk,       Ee, Ee);
  bf16t* pdWv   = packW(dWv,       Ee, Ee);
  bf16t* pdKl   = packW(dWk_logit, Ee, Ee);
  bf16t* pdComb = packW(dcomb_W,   Ee, Ee);
  bf16t* pdWqT  = packW(dWq,       Ee, Ee);        // first 128 rows of dWq
  bf16t* Vt     = wpk + woff;                      // (B, E, TPd) bf16

  auto launch_gemm = [&](const float* A, const bf16t* Wp, const float* bias, float* C,
                         int M, int K, int Nc, int RN, int abTok, int arOff,
                         int cbTok, int crOff, int relu) {
    int totalWaves = (M / 32) * (Nc / 64);
    int blocks = (totalWaves + 7) / 8;
    if (K == 128)
      gemm_kernel<128><<<blocks, 256, 0, stream>>>(A, Wp, bias, C, M, Nc,
                                                   RN, abTok, arOff, cbTok, crOff, relu);
    else
      gemm_kernel<512><<<blocks, 256, 0, stream>>>(A, Wp, bias, C, M, Nc,
                                                   RN, abTok, arOff, cbTok, crOff, relu);
  };
  auto gemm_full = [&](const float* A, const bf16t* Wp, const float* bias, float* C,
                       int M, int K, int Nc, int relu) {
    launch_gemm(A, Wp, bias, C, M, K, Nc, M, 0, 0, 0, 0, relu);
  };
  auto launch_vtrans = [&](const float* V) {
    int n = Bb * Ee * (TPd / 8);
    vtrans_kernel<<<(n + 255) / 256, 256, 0, stream>>>(V, Vt);
  };
  auto launch_attn = [&](const float* Qp, int qbTok, int qOff, int Lq,
                         const float* Kp, int kbTok, int kOff, int Lk,
                         float* Op, int obTok, int oOff, int accF,
                         const float* maskP, int maskB, int maskR) {
    int QT = (Lq + 15) / 16;
    int numWaves = Bb * Hh * QT;
    int blocks = (numWaves + 3) / 4;
    if (Lk == Nn)
      attn_kernel<Nn><<<blocks, 128, 0, stream>>>(Qp, qbTok, qOff, Lq, Kp, kbTok,
                                                  Vt, kOff, Op, obTok, oOff, accF,
                                                  maskP, maskB, maskR, numWaves, QT);
    else
      attn_kernel<Ss><<<blocks, 128, 0, stream>>>(Qp, qbTok, qOff, Lq, Kp, kbTok,
                                                  Vt, kOff, Op, obTok, oOff, accF,
                                                  maskP, maskB, maskR, numWaves, QT);
  };

  // ---- embedding ----
  embed_kernel<<<(NE + 255) / 256, 256, 0, stream>>>(item, sols, emb_W, emb_b,
                                                     semb_W, semb_b, X);

  // ---- encoder layers ----
  for (int i = 0; i < Ll; ++i) {
    gemm_full(X, pWq[i], nullptr, Qb, ROWS, Ee, Ee, 0);
    gemm_full(X, pWk[i], nullptr, Kb, ROWS, Ee, Ee, 0);
    gemm_full(X, pWv[i], nullptr, Vb, ROWS, Ee, Ee, 0);
    launch_vtrans(Vb);

    // oc_nn (=), oc_ns (+=, per-key sols_mask), oc_sn (=)
    launch_attn(Qb, Pp, 0,  Nn, Kb, Pp, 0,  Nn, OC, Pp, 0,  0, nullptr,   0,  0);
    launch_attn(Qb, Pp, 0,  Nn, Kb, Pp, Nn, Ss, OC, Pp, 0,  1, sols_mask, Ss, 0);
    launch_attn(Qb, Pp, Nn, Ss, Kb, Pp, 0,  Nn, OC, Pp, Nn, 0, nullptr,   0,  0);

    launch_gemm(OC, pCo[i], Lcomb_b  + (size_t)i * Ee, MHb,
                Bb * Nn, Ee, Ee, Nn, Pp, 0,  Pp, 0,  0);
    launch_gemm(OC, pCs[i], Lcombs_b + (size_t)i * Ee, MHb,
                Bb * Ss, Ee, Ee, Ss, Pp, Nn, Pp, Nn, 0);

    add_kernel<<<(NE4 + 255) / 256, 256, 0, stream>>>((const float4*)X, (const float4*)MHb,
                                                      (float4*)T1, NE4);
    bnstats_kernel<<<Ee/4, 256, 0, stream>>>(T1, stats, ROWS);
    bnapply_kernel<<<(NE4 + 255) / 256, 256, 0, stream>>>(T1, stats,
        Lbn1_g + (size_t)i * Ee, Lbn1_b + (size_t)i * Ee, NE4);

    gemm_full(T1,  pF1[i], Lff_b1 + (size_t)i * FFf, FFb, ROWS, Ee, FFf, 1);
    gemm_full(FFb, pF2[i], Lff_b2 + (size_t)i * Ee,  T2,  ROWS, FFf, Ee, 0);

    add_kernel<<<(NE4 + 255) / 256, 256, 0, stream>>>((const float4*)T1, (const float4*)T2,
                                                      (float4*)X, NE4);
    bnstats_kernel<<<Ee/4, 256, 0, stream>>>(X, stats, ROWS);
    bnapply_kernel<<<(NE4 + 255) / 256, 256, 0, stream>>>(X, stats,
        Lbn2_g + (size_t)i * Ee, Lbn2_b + (size_t)i * Ee, NE4);
  }

  // ---- decoder ----
  graphmean_kernel<<<(Bb * (Ee/4) + 255) / 256, 256, 0, stream>>>(X, graphB);
  gemm_full(X, pdWk, nullptr, Kb, ROWS, Ee, Ee, 0);
  gemm_full(X, pdWv, nullptr, Vb, ROWS, Ee, Ee, 0);
  launch_vtrans(Vb);
  launch_gemm(X, pdKl, nullptr, shk, Bb * Nn, Ee, Ee, Nn, Pp, 0, Nn, 0, 0);
  gemm_full(graphB, pdWqT, nullptr, qbase, Bb, Ee, Ee, 0);
  qexpand_kernel<<<(Bb * Gg * (Ee/4) + 255) / 256, 256, 0, stream>>>(
      qbase, capacity, dWq + (size_t)Ee * Ee, qdec);

  launch_attn(qdec, Gg, 0, Gg, Kb, Pp, 0,  Nn, ocdec, Gg, 0, 0,
              ninf_mask,      Gg * Nn, Nn);
  launch_attn(qdec, Gg, 0, Gg, Kb, Pp, Nn, Ss, ocdec, Gg, 0, 1,
              sols_mask_pomo, Gg * Ss, Ss);

  gemm_full(ocdec, pdComb, dcomb_b, mhdec, Bb * Gg, Ee, Ee, 0);

  {
    int GT = (Gg + 15) / 16, NT = (Nn + 15) / 16;
    int numWaves = Bb * GT * NT;
    int blocks = (numWaves + 3) / 4;
    score_kernel<<<blocks, 128, 0, stream>>>(mhdec, shk, ninf_mask, scoreB,
                                             numWaves, GT, NT);
  }
  softmax_kernel<<<Bb * Gg, 256, 0, stream>>>(scoreB, (float*)d_out);
}
